// CustomSymplectic_84172769068124
// MI455X (gfx1250) — matrix operand order repaired and checked by reading, not compiled, not run on hardware
//
#include <hip/hip_runtime.h>

typedef __attribute__((ext_vector_type(16))) __bf16   bf16x16;
typedef __attribute__((ext_vector_type(8)))  float    f32x8;
typedef __attribute__((ext_vector_type(4)))  unsigned u32x4;

#define HIDDEN 128
#define NHID   7
#define BTILE  16
// per-net swizzled weight footprint in uint32: 7 layers * 4 ksteps * 8 ntiles * 32 lanes * 8 vgprs
#define NET_STRIDE (7 * 4 * 8 * 256)

__device__ __forceinline__ unsigned short f2bf(float f) {
    unsigned u = __float_as_uint(f);
    u += 0x7FFFu + ((u >> 16) & 1u);          // round-to-nearest-even
    return (unsigned short)(u >> 16);
}
__device__ __forceinline__ float bf2f(unsigned short h) {
    return __uint_as_float(((unsigned)h) << 16);
}
__device__ __forceinline__ void gelu_both(float z, float& g, float& dg) {
    float phi = 0.5f * (1.0f + erff(z * 0.70710678118654752f));  // exact GELU
    float pdf = 0.3989422804014327f * __expf(-0.5f * z * z);
    g  = z * phi;
    dg = phi + z * pdf;
}

union Frag { bf16x16 v; u32x4 q[2]; };

// ---------------------------------------------------------------------------
// Prep: convert hidden-layer weights (fp32, row-major [k][n]) into bf16 packed
// as B-matrix (32x16 bf16) fragments in contiguous k order (k = 16*half + elem).
// Net order: 0=left t0, 1=left t1, 2=right t0, 3=right t1.
// Index: net*NET_STRIDE + ((l*4+kk)*8+n)*256 + lane*8 + v
// Lane L: N = n*16 + (L&15); K = kk*32 + 16*(L>=16) + v*2 (+1 in high half).
// ---------------------------------------------------------------------------
__global__ __launch_bounds__(256) void prep_weights(
    const float* __restrict__ lWh, const float* __restrict__ rWh,
    unsigned* __restrict__ swz)
{
    int idx = blockIdx.x * 256 + threadIdx.x;
    if (idx >= 4 * NET_STRIDE) return;
    int v    = idx & 7;
    int lane = (idx >> 3) & 31;
    int n    = (idx >> 8) & 7;
    int kk   = (idx >> 11) & 3;
    int rest = idx >> 13;          // 0..27
    int l    = rest % 7;
    int net  = rest / 7;

    const float* W = (net < 2 ? lWh + net * NHID * HIDDEN * HIDDEN
                              : rWh + (net - 2) * NHID * HIDDEN * HIDDEN)
                     + l * HIDDEN * HIDDEN;
    int ncol = n * 16 + (lane & 15);
    int k0   = kk * 32 + ((lane >= 16) ? 16 : 0) + v * 2;
    unsigned lo = f2bf(W[k0 * HIDDEN + ncol]);
    unsigned hi = f2bf(W[(k0 + 1) * HIDDEN + ncol]);
    swz[idx] = lo | (hi << 16);
}

// ---------------------------------------------------------------------------
// Fully fused symplectic integrator. One workgroup = 16 batch rows; 8 waves.
// All 7 substeps x (layer0 + 7 WMMA layers + output dot) run in one kernel;
// q/p state lives in LDS, weights stream from L2 via the swizzled buffer.
// ---------------------------------------------------------------------------
__global__ __launch_bounds__(256) void symp_fused(
    const float* __restrict__ X,
    const float* __restrict__ lW0, const float* __restrict__ lb0,
    const float* __restrict__ lbh, const float* __restrict__ lWo,
    const float* __restrict__ rW0, const float* __restrict__ rb0,
    const float* __restrict__ rbh, const float* __restrict__ rWo,
    const unsigned* __restrict__ swz,
    float* __restrict__ out)
{
    __shared__ __align__(16) unsigned short Hs[BTILE * HIDDEN];  // forward activ (bf16)
    __shared__ __align__(16) unsigned short Ts[BTILE * HIDDEN];  // tangent activ (bf16)
    __shared__ float red[BTILE][17];
    __shared__ float xs[BTILE];
    __shared__ float qs[BTILE][2];
    __shared__ float ps[BTILE][2];
    __shared__ float grads[BTILE];

    const int tid  = threadIdx.x;
    const int lane = tid & 31;
    const int wave = tid >> 5;            // 0..7 -> output feature tile
    const int m    = lane & 15;
    const int hi   = (lane >= 16) ? 1 : 0;
    const int row0 = blockIdx.x * BTILE;

    if (tid < BTILE) {
        const float* xr = X + (size_t)(row0 + tid) * 4;
        qs[tid][0] = xr[0]; qs[tid][1] = xr[1];
        ps[tid][0] = xr[2]; ps[tid][1] = xr[3];
    }
    __syncthreads();

    const float dt = 0.1f;
    // Forest-Ruth coefficients: K = 2^(1/3)
    const float Ctab[4] = { 0.67560359597982889f, -0.17560359597982883f,
                           -0.17560359597982883f,  0.67560359597982889f };
    const float Dtab[3] = { 1.35120719195965778f, -1.70241438391931556f,
                            1.35120719195965778f };

    // 7 effective force evaluations: R(C0) L(D0) R(C1) L(D1) R(C2) L(D2) R(C3)
    for (int e = 0; e < 7; ++e) {
        const bool  right = ((e & 1) == 0);
        const float sc    = right ? (Ctab[e >> 1] * dt) : (-Dtab[e >> 1] * dt);
        const float* W0s = right ? rW0 : lW0;
        const float* b0s = right ? rb0 : lb0;
        const float* bhs = right ? rbh : lbh;
        const float* Wos = right ? rWo : lWo;
        const unsigned* swzS = swz + (right ? 2 * NET_STRIDE : 0);

        for (int t = 0; t < 2; ++t) {
            if (tid < BTILE) xs[tid] = right ? ps[tid][t] : qs[tid][t];
            __syncthreads();

            const float* W0n = W0s + t * HIDDEN;
            const float* b0n = b0s + t * HIDDEN;
            const float* bhn = bhs + t * NHID * HIDDEN;
            const float* Won = Wos + t * HIDDEN;
            const unsigned* swzN = swzS + t * NET_STRIDE;

            // ---- layer 0: scalar input -> 128, plus tangent seed ----
            {
                int r  = tid >> 4;
                int cb = (tid & 15) * 8;
                float xv = xs[r];
                #pragma unroll
                for (int j = 0; j < 8; ++j) {
                    int c = cb + j;
                    float w0 = W0n[c];
                    float z  = xv * w0 + b0n[c];
                    float g, dg; gelu_both(z, g, dg);
                    Hs[r * HIDDEN + c] = f2bf(g);
                    Ts[r * HIDDEN + c] = f2bf(w0 * dg);   // dz/dx = w0
                }
            }
            __syncthreads();

            // ---- 7 hidden layers: dual WMMA GEMM (forward + tangent) ----
            for (int l = 0; l < NHID; ++l) {
                f32x8 ah = {0,0,0,0,0,0,0,0};
                f32x8 at = {0,0,0,0,0,0,0,0};
                #pragma unroll
                for (int kk = 0; kk < 4; ++kk) {
                    Frag fh, ft, fb;
                    int colA = kk * 32 + (hi ? 16 : 0);
                    fh.q[0] = *(const u32x4*)&Hs[m * HIDDEN + colA];
                    fh.q[1] = *(const u32x4*)&Hs[m * HIDDEN + colA + 8];
                    ft.q[0] = *(const u32x4*)&Ts[m * HIDDEN + colA];
                    ft.q[1] = *(const u32x4*)&Ts[m * HIDDEN + colA + 8];
                    const u32x4* pb = (const u32x4*)(swzN + ((((l * 4 + kk) * 8 + wave) << 8) + (lane << 3)));
                    fb.q[0] = pb[0]; fb.q[1] = pb[1];
                    ah = __builtin_amdgcn_wmma_f32_16x16x32_bf16(
                             false, fh.v, false, fb.v, (short)0, ah, false, false);
                    at = __builtin_amdgcn_wmma_f32_16x16x32_bf16(
                             false, ft.v, false, fb.v, (short)0, at, false, false);
                }
                __syncthreads();   // all reads of Hs/Ts done before overwrite

                float bias = bhn[l * HIDDEN + wave * 16 + m];
                #pragma unroll
                for (int ei = 0; ei < 8; ++ei) {
                    float z = ah[ei] + bias;
                    float g, dg; gelu_both(z, g, dg);
                    int M = ei + (hi ? 8 : 0);
                    int N = wave * 16 + m;
                    Hs[M * HIDDEN + N] = f2bf(g);
                    Ts[M * HIDDEN + N] = f2bf(at[ei] * dg);
                }
                __syncthreads();
            }

            // ---- output: grad[r] = Ts[r,:] . Wo  (bias drops in derivative) ----
            {
                int r  = tid >> 4;
                int cb = (tid & 15) * 8;
                float s = 0.0f;
                #pragma unroll
                for (int j = 0; j < 8; ++j) {
                    int c = cb + j;
                    s += bf2f(Ts[r * HIDDEN + c]) * Won[c];
                }
                red[r][tid & 15] = s;
            }
            __syncthreads();
            if (tid < BTILE) {
                float g = 0.0f;
                #pragma unroll
                for (int j = 0; j < 16; ++j) g += red[tid][j];
                grads[tid] = g;
            }
            __syncthreads();

            if (tid < BTILE) {
                if (right) qs[tid][t] += sc * grads[tid];
                else       ps[tid][t] += sc * grads[tid];
            }
            __syncthreads();
        }
    }

    if (tid < BTILE) {
        float* o = out + (size_t)(row0 + tid) * 4;
        o[0] = qs[tid][0]; o[1] = qs[tid][1];
        o[2] = ps[tid][0]; o[3] = ps[tid][1];
    }
}

extern "C" void kernel_launch(void* const* d_in, const int* in_sizes, int n_in,
                              void* d_out, int out_size, void* d_ws, size_t ws_size,
                              hipStream_t stream) {
    (void)in_sizes; (void)n_in; (void)out_size; (void)ws_size;
    const float* X   = (const float*)d_in[0];
    const float* lW0 = (const float*)d_in[1];
    const float* lb0 = (const float*)d_in[2];
    const float* lWh = (const float*)d_in[3];
    const float* lbh = (const float*)d_in[4];
    const float* lWo = (const float*)d_in[5];
    const float* rW0 = (const float*)d_in[7];
    const float* rb0 = (const float*)d_in[8];
    const float* rWh = (const float*)d_in[9];
    const float* rbh = (const float*)d_in[10];
    const float* rWo = (const float*)d_in[11];

    unsigned* swz = (unsigned*)d_ws;   // 4*NET_STRIDE*4 bytes = ~0.9 MB

    prep_weights<<<(4 * NET_STRIDE + 255) / 256, 256, 0, stream>>>(lWh, rWh, swz);
    symp_fused<<<16384 / BTILE, 256, 0, stream>>>(
        X, lW0, lb0, lbh, lWo, rW0, rb0, rbh, rWo, swz, (float*)d_out);
}
